// StructureConditionedAttention_56667798503540
// MI455X (gfx1250) — matrix operand order repaired and checked by reading, not compiled, not run on hardware
//
#include <hip/hip_runtime.h>
#include <hip/hip_bf16.h>

// ---------------------------------------------------------------------------
// Problem constants (StructureConditionedAttention):
//   B=4, E=2048, D=256, H=8, DH=32 ; M = B*E = 8192 rows
// All GEMMs run on v_wmma_f32_16x16x32_bf16 (f32 accumulate).
// ---------------------------------------------------------------------------
#define BB 4
#define EE 2048
#define DD 256
#define HH 8
#define DH 32
#define MM (BB * EE)   // 8192

typedef __bf16 bf16_t;
typedef __attribute__((ext_vector_type(16))) __bf16 v16bf;
typedef __attribute__((ext_vector_type(8)))  float  v8f;
typedef __attribute__((__vector_size__(4 * sizeof(int)))) int vi4;  // builtin param type

// ---- CDNA5 async global->LDS path (guarded; sync LDS-store fallback) ------
#if defined(__has_builtin)
#  if __has_builtin(__builtin_amdgcn_global_load_async_to_lds_b128)
#    define USE_ASYNC_LDS 1
#  endif
#  if __has_builtin(__builtin_amdgcn_s_wait_asynccnt)
#    define WAIT_ASYNC(n) __builtin_amdgcn_s_wait_asynccnt(n)
#  else
#    define WAIT_ASYNC(n) asm volatile("s_wait_asynccnt %0" ::"i"(n) : "memory")
#  endif
#endif

// ---------------------------------------------------------------------------
// WMMA fragment helpers (layouts per CDNA5 ISA 7.12.2, wave32)
//
// A (16x32 bf16, M x K), A row-major:
//   lane l (l<16): row M=l,    K = {k0+0..7}  and {k0+16..23}
//   lane l (>=16): row M=l-16, K = {k0+8..15} and {k0+24..31}
// B (32x16 bf16, K x N), stored as BT row-major (N x K):
//   lane l (l<16): col N=l,    K = {k0+0..7}  and {k0+16..23}
//   lane l (>=16): col N=l-16, K = {k0+8..15} and {k0+24..31}
// C/D (16x16 f32): lane l holds col N=(l&15), rows M=(l>>4)*8 + r.
// ---------------------------------------------------------------------------
static __device__ __forceinline__ v16bf load_frag_a(const bf16_t* __restrict__ A,
                                                    int lda, int row0, int k0) {
  const int l    = threadIdx.x & 31;
  const int half = l >> 4;
  const int r    = l & 15;
  const bf16_t* p = A + (size_t)(row0 + r) * lda + k0 + half * 8;
  union { v16bf v; uint4 q[2]; } u;
  u.q[0] = *reinterpret_cast<const uint4*>(p);
  u.q[1] = *reinterpret_cast<const uint4*>(p + 16);
  return u.v;
}

static __device__ __forceinline__ v16bf load_frag_b(const bf16_t* __restrict__ BT,
                                                    int ldb, int col0, int k0) {
  const int l    = threadIdx.x & 31;
  const int half = l >> 4;
  const int n    = l & 15;
  const bf16_t* p = BT + (size_t)(col0 + n) * ldb + k0 + half * 8;
  union { v16bf v; uint4 q[2]; } u;
  u.q[0] = *reinterpret_cast<const uint4*>(p);
  u.q[1] = *reinterpret_cast<const uint4*>(p + 16);
  return u.v;
}

static __device__ __forceinline__ v8f wmma_bf16(v16bf a, v16bf b, v8f c) {
  return __builtin_amdgcn_wmma_f32_16x16x32_bf16(false, a, false, b,
                                                 (short)0, c, false, false);
}

// ---------------------------------------------------------------------------
// Shared GEMM core: block = 256 thr (8 waves) -> 128(M) x 64(N) tile.
// Per wave: 16(M) x 64(N), i.e. 4 accumulators sharing one A fragment.
// B tile (64 N x 64 K bf16 = 8KB) is staged into LDS (double-buffered,
// async global->LDS when available) and shared by all 8 waves.
// ---------------------------------------------------------------------------
static __device__ __forceinline__ void gemm_core_16x64(
    const bf16_t* __restrict__ A, const bf16_t* __restrict__ BT,
    int Kdim, int row0, int colb, bf16_t (*Bl)[64 * 64], v8f acc[4]) {
  const int t    = threadIdx.x;
  const int srow = t >> 2;          // 0..63 : N row of staged tile
  const int skk  = (t & 3) * 16;    // element offset within 64-wide K chunk

  auto stage = [&](int k0, int buf) {
    const bf16_t* gp = BT + (size_t)(colb + srow) * Kdim + k0 + skk;
    bf16_t* lp = &Bl[buf][srow * 64 + skk];
#ifdef USE_ASYNC_LDS
    __builtin_amdgcn_global_load_async_to_lds_b128((vi4*)gp, (vi4*)lp, 0, 0);
    __builtin_amdgcn_global_load_async_to_lds_b128((vi4*)(gp + 8), (vi4*)(lp + 8), 0, 0);
#else
    *reinterpret_cast<uint4*>(lp)     = *reinterpret_cast<const uint4*>(gp);
    *reinterpret_cast<uint4*>(lp + 8) = *reinterpret_cast<const uint4*>(gp + 8);
#endif
  };

  const int nchunks = Kdim >> 6;    // K / 64
  stage(0, 0);
  for (int c = 0; c < nchunks; ++c) {
    const int k0  = c << 6;
    const int buf = c & 1;
    if (c + 1 < nchunks) {
      stage(k0 + 64, buf ^ 1);
      __builtin_prefetch(A + (size_t)row0 * Kdim + k0 + 64, 0, 1);
#ifdef USE_ASYNC_LDS
      WAIT_ASYNC(2);                // chunk c complete, c+1 may be in flight
#endif
    } else {
#ifdef USE_ASYNC_LDS
      WAIT_ASYNC(0);
#endif
    }
    __syncthreads();                // all waves' stage of chunk c visible
#pragma unroll
    for (int kk = 0; kk < 64; kk += 32) {
      v16bf a = load_frag_a(A, Kdim, row0, k0 + kk);
#pragma unroll
      for (int j = 0; j < 4; ++j) {
        v16bf b = load_frag_b(&Bl[buf][0], 64, j * 16, kk);
        acc[j] = wmma_bf16(a, b, acc[j]);
      }
    }
    __syncthreads();                // buffer reusable two chunks later
  }
}

// ---------------------------------------------------------------------------
// Kernel: f32 -> bf16 straight convert (x -> Xbf)
// ---------------------------------------------------------------------------
__global__ void cvt_f32_bf16_kernel(const float* __restrict__ in,
                                    bf16_t* __restrict__ out, int n) {
  int i = blockIdx.x * blockDim.x + threadIdx.x;
  if (i < n) out[i] = (bf16_t)in[i];
}

// ---------------------------------------------------------------------------
// Kernel: transpose + convert W (R x C f32, in-major) -> out (C x R bf16)
// ---------------------------------------------------------------------------
__global__ void transpose_cvt_kernel(const float* __restrict__ W,
                                     bf16_t* __restrict__ out, int R, int C) {
  int i = blockIdx.x * blockDim.x + threadIdx.x;
  if (i < R * C) {
    int r = i / C, c = i % C;
    out[(size_t)c * R + r] = (bf16_t)W[i];
  }
}

// ---------------------------------------------------------------------------
// Kernel: fused QKV projection. M=8192, K=256, N=768 (WqT|WkT|WvT).
//   Q : (B,H,E,DH) bf16   K : (B,H,E,DH) bf16   Vt : (B,H,DH,E) bf16
// grid = (768/64, M/128), block = 256.
// ---------------------------------------------------------------------------
__global__ void qkv_gemm_kernel(const bf16_t* __restrict__ Xbf,
                                const bf16_t* __restrict__ WqkvT,
                                const float* __restrict__ bq,
                                const float* __restrict__ bk,
                                const float* __restrict__ bv,
                                bf16_t* __restrict__ Qo,
                                bf16_t* __restrict__ Ko,
                                bf16_t* __restrict__ Vt) {
  __shared__ bf16_t Bl[2][64 * 64];
  const int wave = threadIdx.x >> 5;
  const int row0 = blockIdx.y * 128 + wave * 16;
  const int colb = blockIdx.x * 64;

  v8f acc[4] = {};
  gemm_core_16x64(Xbf, WqkvT, DD, row0, colb, Bl, acc);

  const int l     = threadIdx.x & 31;
  const int rbase = row0 + (l >> 4) * 8;
#pragma unroll
  for (int j = 0; j < 4; ++j) {
    const int n      = colb + j * 16 + (l & 15);
    const int which  = n >> 8;          // 0=q 1=k 2=v
    const int within = n & 255;
    const int h      = within >> 5;
    const int dh     = within & 31;
    const float bias = (which == 0) ? bq[within]
                     : (which == 1) ? bk[within] : bv[within];
#pragma unroll
    for (int r = 0; r < 8; ++r) {
      const int m  = rbase + r;
      const int bi = m >> 11;
      const int e  = m & (EE - 1);
      const float val = acc[j][r] + bias;
      if (which == 0)
        Qo[((((size_t)bi * HH + h) * EE + e) * DH) + dh] = (bf16_t)val;
      else if (which == 1)
        Ko[((((size_t)bi * HH + h) * EE + e) * DH) + dh] = (bf16_t)val;
      else
        Vt[((((size_t)bi * HH + h) * DH + dh) * EE) + e] = (bf16_t)val;
    }
  }
}

// ---------------------------------------------------------------------------
// Kernel: flash attention with structure bias.
// grid = (E/16, H, B), block = 32 (one wave per 16-query tile).
// DH=32 == WMMA K: Q@K^T per 16x16 tile is ONE v_wmma.
// P (16x32) transposed C-layout -> A-layout through LDS.
// ---------------------------------------------------------------------------
__global__ void attn_kernel(const bf16_t* __restrict__ Q,
                            const bf16_t* __restrict__ K,
                            const bf16_t* __restrict__ Vt,
                            const float* __restrict__ SB,
                            const unsigned char* __restrict__ kpm,
                            const float* __restrict__ bias_scale_p,
                            bf16_t* __restrict__ X2) {
  const int qt = blockIdx.x;
  const int h  = blockIdx.y;
  const int b  = blockIdx.z;
  const int l  = threadIdx.x & 31;
  const int col  = l & 15;
  const int half = l >> 4;

  const bf16_t* Qbh = Q  + (((size_t)b * HH + h) * EE) * DH;
  const bf16_t* Kbh = K  + (((size_t)b * HH + h) * EE) * DH;
  const bf16_t* Vbh = Vt + (((size_t)b * HH + h) * DH) * EE;
  const float scale = 0.17677669529663687f;     // 1/sqrt(32)
  const float bsc   = bias_scale_p[0];

  const v16bf aq = load_frag_a(Qbh, DH, qt * 16, 0);

  __shared__ bf16_t Pl[16 * 32];

  v8f o0 = {}, o1 = {};
  float mrow[8], srow[8];
#pragma unroll
  for (int r = 0; r < 8; ++r) { mrow[r] = -3.0e38f; srow[r] = 0.f; }

  const int q0 = qt * 16 + half * 8;

  for (int kb = 0; kb < EE; kb += 32) {
    v16bf b0 = load_frag_b(Kbh, DH, kb, 0);
    v16bf b1 = load_frag_b(Kbh, DH, kb + 16, 0);
    v8f z = {};
    v8f s0 = wmma_bf16(aq, b0, z);
    v8f s1 = wmma_bf16(aq, b1, z);

    const int key0 = kb + col;
    const int key1 = kb + 16 + col;
    const bool msk0 = kpm[(size_t)b * EE + key0] != 0;
    const bool msk1 = kpm[(size_t)b * EE + key1] != 0;

    float t0[8], t1[8];
#pragma unroll
    for (int r = 0; r < 8; ++r) {
      const int q = q0 + r;
      const size_t sbrow = ((size_t)b * EE + q) * EE;
      float v0 = s0[r] * scale + bsc * SB[sbrow + key0];
      float v1 = s1[r] * scale + bsc * SB[sbrow + key1];
      if (msk0) v0 = -3.0e38f;
      if (msk1) v1 = -3.0e38f;
      t0[r] = v0; t1[r] = v1;
    }

    float bm[8];
#pragma unroll
    for (int r = 0; r < 8; ++r) bm[r] = fmaxf(t0[r], t1[r]);
#pragma unroll
    for (int m = 1; m < 16; m <<= 1)
#pragma unroll
      for (int r = 0; r < 8; ++r) bm[r] = fmaxf(bm[r], __shfl_xor(bm[r], m));

#pragma unroll
    for (int r = 0; r < 8; ++r) {
      const float mnew  = fmaxf(mrow[r], bm[r]);
      const float alpha = __expf(mrow[r] - mnew);
      mrow[r] = mnew;
      const float p0 = __expf(t0[r] - mnew);
      const float p1 = __expf(t1[r] - mnew);
      srow[r] = srow[r] * alpha + p0 + p1;
      o0[r] *= alpha; o1[r] *= alpha;
      t0[r] = p0; t1[r] = p1;
    }

    __syncthreads();
#pragma unroll
    for (int r = 0; r < 8; ++r) {
      Pl[(half * 8 + r) * 32 + col]      = (bf16_t)t0[r];
      Pl[(half * 8 + r) * 32 + col + 16] = (bf16_t)t1[r];
    }
    __syncthreads();
    const v16bf ap = load_frag_a(Pl, 32, 0, 0);

    v16bf bv0 = load_frag_b(Vbh, EE, 0, kb);
    v16bf bv1 = load_frag_b(Vbh, EE, 16, kb);
    o0 = wmma_bf16(ap, bv0, o0);
    o1 = wmma_bf16(ap, bv1, o1);
  }

#pragma unroll
  for (int m = 1; m < 16; m <<= 1)
#pragma unroll
    for (int r = 0; r < 8; ++r) srow[r] += __shfl_xor(srow[r], m);

#pragma unroll
  for (int r = 0; r < 8; ++r) {
    const float inv = 1.0f / srow[r];
    const int e = q0 + r;
    bf16_t* dst = X2 + ((size_t)b * EE + e) * DD + h * DH;
    dst[col]      = (bf16_t)(o0[r] * inv);
    dst[col + 16] = (bf16_t)(o1[r] * inv);
  }
}

// ---------------------------------------------------------------------------
// Kernel: generic bf16 WMMA GEMM, C = A(MxK) @ BT^T + bias, optional exact
// GELU / f32 residual add / f32 and-or bf16 outputs.
// grid = (N/64, M/128), block = 256.
// ---------------------------------------------------------------------------
__global__ void gemm_bf16_kernel(const bf16_t* __restrict__ A,
                                 const bf16_t* __restrict__ BT,
                                 const float* __restrict__ bias,
                                 const float* __restrict__ residual,
                                 float* __restrict__ outF,
                                 bf16_t* __restrict__ outB,
                                 int M, int N, int Kdim, int gelu) {
  __shared__ bf16_t Bl[2][64 * 64];
  const int wave = threadIdx.x >> 5;
  const int row0 = blockIdx.y * 128 + wave * 16;
  const int colb = blockIdx.x * 64;

  v8f acc[4] = {};
  gemm_core_16x64(A, BT, Kdim, row0, colb, Bl, acc);

  const int l     = threadIdx.x & 31;
  const int rbase = row0 + (l >> 4) * 8;
#pragma unroll
  for (int j = 0; j < 4; ++j) {
    const int n    = colb + j * 16 + (l & 15);
    const float bv = bias ? bias[n] : 0.f;
#pragma unroll
    for (int r = 0; r < 8; ++r) {
      float v = acc[j][r] + bv;
      if (gelu) v = 0.5f * v * (1.0f + erff(v * 0.70710678118654752f));
      const size_t idx = (size_t)(rbase + r) * N + n;
      if (residual) v += residual[idx];
      if (outF) outF[idx] = v;
      if (outB) outB[idx] = (bf16_t)v;
    }
  }
}

// ---------------------------------------------------------------------------
// Kernel: LayerNorm over D=256. One 256-thread block per row.
// ---------------------------------------------------------------------------
__global__ void layernorm_kernel(const float* __restrict__ in,
                                 const float* __restrict__ g,
                                 const float* __restrict__ be,
                                 float* __restrict__ outF,
                                 bf16_t* __restrict__ outB) {
  const int row = blockIdx.x;
  const int t   = threadIdx.x;
  const float v = in[(size_t)row * DD + t];

  __shared__ float red[8];
  float s = v;
#pragma unroll
  for (int m = 1; m < 32; m <<= 1) s += __shfl_xor(s, m);
  if ((t & 31) == 0) red[t >> 5] = s;
  __syncthreads();
  float tot = 0.f;
#pragma unroll
  for (int i = 0; i < 8; ++i) tot += red[i];
  const float mean = tot * (1.0f / DD);
  __syncthreads();

  const float d = v - mean;
  float s2 = d * d;
#pragma unroll
  for (int m = 1; m < 32; m <<= 1) s2 += __shfl_xor(s2, m);
  if ((t & 31) == 0) red[t >> 5] = s2;
  __syncthreads();
  float tot2 = 0.f;
#pragma unroll
  for (int i = 0; i < 8; ++i) tot2 += red[i];
  const float var = tot2 * (1.0f / DD);

  const float y = d * rsqrtf(var + 1e-5f) * g[t] + be[t];
  const size_t idx = (size_t)row * DD + t;
  if (outF) outF[idx] = y;
  if (outB) outB[idx] = (bf16_t)y;
}

// ---------------------------------------------------------------------------
// Host launcher
// ---------------------------------------------------------------------------
extern "C" void kernel_launch(void* const* d_in, const int* in_sizes, int n_in,
                              void* d_out, int out_size, void* d_ws, size_t ws_size,
                              hipStream_t stream) {
  (void)in_sizes; (void)n_in; (void)out_size; (void)ws_size;

  const float* x   = (const float*)d_in[0];
  const float* SB  = (const float*)d_in[1];
  const unsigned char* kpm = (const unsigned char*)d_in[2];
  const float* Wq  = (const float*)d_in[3];
  const float* bq  = (const float*)d_in[4];
  const float* Wk  = (const float*)d_in[5];
  const float* bk  = (const float*)d_in[6];
  const float* Wv  = (const float*)d_in[7];
  const float* bv  = (const float*)d_in[8];
  const float* Wo  = (const float*)d_in[9];
  const float* bo  = (const float*)d_in[10];
  const float* bias_scale = (const float*)d_in[11];
  const float* g1  = (const float*)d_in[12];
  const float* b1  = (const float*)d_in[13];
  const float* W1  = (const float*)d_in[14];
  const float* bf1 = (const float*)d_in[15];
  const float* W2  = (const float*)d_in[16];
  const float* bf2 = (const float*)d_in[17];
  const float* g2  = (const float*)d_in[18];
  const float* b2  = (const float*)d_in[19];

  char* ws = (char*)d_ws;
  size_t off = 0;
  auto take = [&](size_t bytes) -> void* {
    void* p = ws + off;
    off += (bytes + 255) & ~(size_t)255;
    return p;
  };
  bf16_t* Xbf   = (bf16_t*)take((size_t)MM * DD * 2);
  bf16_t* WqkvT = (bf16_t*)take((size_t)3 * DD * DD * 2);
  bf16_t* WoT   = (bf16_t*)take((size_t)DD * DD * 2);
  bf16_t* W1T   = (bf16_t*)take((size_t)4 * DD * DD * 2);
  bf16_t* W2T   = (bf16_t*)take((size_t)4 * DD * DD * 2);
  bf16_t* Qb    = (bf16_t*)take((size_t)MM * DD * 2);
  bf16_t* Kb    = (bf16_t*)take((size_t)MM * DD * 2);
  bf16_t* Vt    = (bf16_t*)take((size_t)MM * DD * 2);
  bf16_t* X2    = (bf16_t*)take((size_t)MM * DD * 2);
  float*  y1    = (float*)take((size_t)MM * DD * 4);
  float*  h1f   = (float*)take((size_t)MM * DD * 4);
  bf16_t* h1b   = (bf16_t*)take((size_t)MM * DD * 2);
  bf16_t* ffb   = (bf16_t*)take((size_t)MM * 4 * DD * 2);
  float*  y2    = (float*)take((size_t)MM * DD * 4);

  // 0) conversions + weight transposes
  {
    int n = MM * DD;
    cvt_f32_bf16_kernel<<<(n + 255) / 256, 256, 0, stream>>>(x, Xbf, n);
  }
  {
    int n = DD * DD;
    transpose_cvt_kernel<<<(n + 255) / 256, 256, 0, stream>>>(Wq, WqkvT,           DD, DD);
    transpose_cvt_kernel<<<(n + 255) / 256, 256, 0, stream>>>(Wk, WqkvT + DD * DD, DD, DD);
    transpose_cvt_kernel<<<(n + 255) / 256, 256, 0, stream>>>(Wv, WqkvT + 2*DD*DD, DD, DD);
    transpose_cvt_kernel<<<(n + 255) / 256, 256, 0, stream>>>(Wo, WoT,             DD, DD);
    int n1 = DD * 4 * DD;
    transpose_cvt_kernel<<<(n1 + 255) / 256, 256, 0, stream>>>(W1, W1T, DD, 4 * DD);
    transpose_cvt_kernel<<<(n1 + 255) / 256, 256, 0, stream>>>(W2, W2T, 4 * DD, DD);
  }

  // 1) fused QKV projection
  {
    dim3 grid(3 * DD / 64, MM / 128);
    qkv_gemm_kernel<<<grid, 256, 0, stream>>>(Xbf, WqkvT, bq, bk, bv, Qb, Kb, Vt);
  }

  // 2) flash attention with structure bias
  {
    dim3 grid(EE / 16, HH, BB);
    attn_kernel<<<grid, 32, 0, stream>>>(Qb, Kb, Vt, SB, kpm, bias_scale, X2);
  }

  // 3) out projection + residual(x) -> y1 (f32)
  {
    dim3 grid(DD / 64, MM / 128);
    gemm_bf16_kernel<<<grid, 256, 0, stream>>>(X2, WoT, bo, x, y1, nullptr,
                                               MM, DD, DD, 0);
  }

  // 4) LayerNorm1 -> h1 (f32 + bf16)
  layernorm_kernel<<<MM, DD, 0, stream>>>(y1, g1, b1, h1f, h1b);

  // 5) FFN up + exact GELU -> bf16
  {
    dim3 grid(4 * DD / 64, MM / 128);
    gemm_bf16_kernel<<<grid, 256, 0, stream>>>(h1b, W1T, bf1, nullptr, nullptr, ffb,
                                               MM, 4 * DD, DD, 1);
  }

  // 6) FFN down + residual(h1) -> y2 (f32)
  {
    dim3 grid(DD / 64, MM / 128);
    gemm_bf16_kernel<<<grid, 256, 0, stream>>>(ffb, W2T, bf2, h1f, y2, nullptr,
                                               MM, DD, 4 * DD, 0);
  }

  // 7) LayerNorm2 -> final output (f32)
  layernorm_kernel<<<MM, DD, 0, stream>>>(y2, g2, b2, (float*)d_out, nullptr);
}
